// OutputLayer_31945966747724
// MI455X (gfx1250) — matrix operand order repaired and checked by reading, not compiled, not run on hardware
//
#include <hip/hip_runtime.h>
#include <hip/hip_bf16.h>

// ---------------------------------------------------------------------------
// Fused label-wise-attention output layer for MI455X (gfx1250, wave32, WMMA).
//
//   scores = U_w @ x^T          [B,Y,L]   (bf16 WMMA, f32 accum)
//   alpha  = softmax_L(scores)            (online / flash-style)
//   m      = alpha @ x          [B,Y,D]   (bf16 WMMA, f32 accum)
//   y      = sum_d(final_w * m) + final_b [B,Y]
//
// v2: bf16 pre-conversion pass into d_ws (row-major + transposed x),
//     double-buffered chunk staging via GLOBAL_LOAD_ASYNC_TO_LDS_B128 +
//     s_wait_asynccnt, U tile held in 128 persistent VGPRs (A-fragments),
//     softmax parallelized 4 threads/row.
// Workspace requirement: ~30.2 MB in d_ws.
// ---------------------------------------------------------------------------

typedef __bf16 bf16;
typedef __attribute__((ext_vector_type(16))) __bf16 v16bf;
typedef __attribute__((ext_vector_type(8)))  float  v8f;
typedef int v4i __attribute__((vector_size(16)));
typedef __attribute__((address_space(1))) v4i  gv4i;   // global int4
typedef __attribute__((address_space(3))) v4i  lv4i;   // LDS int4
typedef __attribute__((address_space(1))) void gvoid;
typedef __attribute__((address_space(3))) void lvoid;

// ---- problem constants ----------------------------------------------------
#define CB   4
#define CL   2500
#define CD   512
#define CY   8921
#define LPAD 2560            // 40 chunks of 64, zero padded
#define YPAD 8960            // 140 tiles of 64, zero padded
#define YT   64
#define LC   64
#define XRLD 520             // padded LDS row strides (bank-conflict free frags)
#define XCLD 72
#define SLD  68
#define PLD  72

#if __has_builtin(__builtin_amdgcn_global_load_async_to_lds_b128) && \
    __has_builtin(__builtin_amdgcn_s_wait_asynccnt)
#define USE_ASYNC 1
#else
#define USE_ASYNC 0
#endif

#if USE_ASYNC
#define WAIT_ASYNC(n) __builtin_amdgcn_s_wait_asynccnt(n)
#else
#define WAIT_ASYNC(n)
#endif

__device__ __forceinline__ void async_copy_b128(const bf16* g, bf16* l) {
#if USE_ASYNC
  __builtin_amdgcn_global_load_async_to_lds_b128(
      (gv4i*)(gvoid*)(bf16*)g,
      (lv4i*)(lvoid*)l,
      /*offset=*/0, /*cpol=*/0);
#else
  *(uint4*)l = *(const uint4*)g;
#endif
}

__device__ __forceinline__ bf16 f2bf(float f) {
  union { float f; unsigned u; } a; a.f = f;
  unsigned r = a.u + 0x7FFFu + ((a.u >> 16) & 1u);   // round-to-nearest-even
  union { unsigned short s; bf16 b; } c;
  c.s = (unsigned short)(r >> 16);
  return c.b;
}

__device__ __forceinline__ unsigned packbf2(float a, float b) {
  union { bf16 h[2]; unsigned u; } t;
  t.h[0] = f2bf(a); t.h[1] = f2bf(b);
  return t.u;
}

// A fragment (16x32 bf16, row-major source): lane m=lane&15, h=lane>>4 holds
// k in [k0+8h, k0+8h+8) and [k0+16+8h, k0+16+8h+8)  (two 16B chunks).
__device__ __forceinline__ v16bf loadA(const bf16* base, int ldm, int row0,
                                       int k0, int lane) {
  const int m = lane & 15, h = lane >> 4;
  const bf16* r = base + (row0 + m) * ldm;
  union { uint4 u[2]; v16bf v; } t;
  t.u[0] = *(const uint4*)(r + k0 + 8 * h);
  t.u[1] = *(const uint4*)(r + k0 + 16 + 8 * h);
  return t.v;
}

// B fragment (32x16 bf16): lane n=lane&15, h=lane>>4 holds k in
// [k0+8h, k0+8h+8) and [k0+16+8h, k0+16+8h+8) for column n, source stored n-major.
__device__ __forceinline__ v16bf loadB(const bf16* base, int ldm, int n0,
                                       int k0, int lane) {
  const int n = lane & 15, h = lane >> 4;
  const bf16* r = base + (n0 + n) * ldm + k0;
  union { uint4 u[2]; v16bf v; } t;
  t.u[0] = *(const uint4*)(r + 8 * h);
  t.u[1] = *(const uint4*)(r + 16 + 8 * h);
  return t.v;
}

__device__ __forceinline__ v8f zero8() {
  v8f z;
#pragma unroll
  for (int j = 0; j < 8; ++j) z[j] = 0.0f;
  return z;
}

// ---------------------------------------------------------------------------
// prep kernels: fp32 -> bf16, zero-padded, row-major + transposed x
// ---------------------------------------------------------------------------
__global__ __launch_bounds__(256) void prep_U(const float* __restrict__ U_w,
                                              bf16* __restrict__ Ub) {
  const int i4 = blockIdx.x * 256 + threadIdx.x;   // grid covers YPAD*CD/4
  const int e = i4 * 4;
  const int y = e >> 9, c = e & 511;
  float4 v = make_float4(0.f, 0.f, 0.f, 0.f);
  if (y < CY) v = *(const float4*)(U_w + y * CD + c);
  uint2 o; o.x = packbf2(v.x, v.y); o.y = packbf2(v.z, v.w);
  *(uint2*)(Ub + e) = o;
}

__global__ __launch_bounds__(256) void prep_x(const float* __restrict__ x,
                                              bf16* __restrict__ Xr,
                                              bf16* __restrict__ Xt) {
  __shared__ float tile[32][36];
  const int tid = threadIdx.x;
  const int b = blockIdx.z;
  const int lt = blockIdx.x * 32;
  const int dt = blockIdx.y * 32;
  // phase 1: read 32(l) x 32(d) fp32 tile, emit row-major bf16
  {
    const int tx4 = tid & 7, ty = tid >> 3;          // d-quad, l-row
    const int l = lt + ty, d = dt + tx4 * 4;
    float4 v = make_float4(0.f, 0.f, 0.f, 0.f);
    if (l < CL) v = *(const float4*)(x + ((size_t)b * CL + l) * CD + d);
    uint2 o; o.x = packbf2(v.x, v.y); o.y = packbf2(v.z, v.w);
    *(uint2*)(Xr + ((size_t)b * LPAD + l) * CD + d) = o;
    tile[ty][tx4 * 4 + 0] = v.x;
    tile[ty][tx4 * 4 + 1] = v.y;
    tile[ty][tx4 * 4 + 2] = v.z;
    tile[ty][tx4 * 4 + 3] = v.w;
  }
  __syncthreads();
  // phase 2: emit transposed bf16, coalesced along l
  {
    const int lx4 = tid & 7, dy = tid >> 3;          // l-quad, d-row
    const int l = lt + lx4 * 4, d = dt + dy;
    const float a0 = tile[lx4 * 4 + 0][dy];
    const float a1 = tile[lx4 * 4 + 1][dy];
    const float a2 = tile[lx4 * 4 + 2][dy];
    const float a3 = tile[lx4 * 4 + 3][dy];
    uint2 o; o.x = packbf2(a0, a1); o.y = packbf2(a2, a3);
    *(uint2*)(Xt + ((size_t)b * CD + d) * LPAD + l) = o;
  }
}

// ---------------------------------------------------------------------------
// issue one chunk's staging: 64x512 row-major + 512x64 transposed, 16B/lane
// chunks; 32 async b128 per thread.
// ---------------------------------------------------------------------------
__device__ __forceinline__ void stage_chunk(const bf16* __restrict__ gr,
                                            const bf16* __restrict__ gt,
                                            int l0, bf16* lr, bf16* lc,
                                            int tid) {
#pragma unroll
  for (int j = 0; j < 16; ++j) {            // sXr: 64 rows x 64 chunks
    const int c = tid + 256 * j;
    const int r = c >> 6, q = c & 63;
    async_copy_b128(gr + (size_t)(l0 + r) * CD + q * 8, lr + r * XRLD + q * 8);
  }
#pragma unroll
  for (int j = 0; j < 16; ++j) {            // sXc: 512 rows x 8 chunks
    const int c = tid + 256 * j;
    const int d = c >> 3, q = c & 7;
    async_copy_b128(gt + (size_t)d * LPAD + l0 + q * 8, lc + d * XCLD + q * 8);
  }
}

// ---------------------------------------------------------------------------
// main fused kernel
// ---------------------------------------------------------------------------
__global__ __launch_bounds__(256) void laat_main(
    const bf16* __restrict__ Ub,       // [YPAD, D] bf16 (zero-padded rows)
    const bf16* __restrict__ Xr,       // [B, LPAD, D] bf16 (zero-padded)
    const bf16* __restrict__ Xt,       // [B, D, LPAD] bf16 (zero-padded)
    const float* __restrict__ final_w, // [Y, D] fp32
    const float* __restrict__ final_b, // [Y]
    float* __restrict__ out)           // [B, Y]
{
  __shared__ alignas(16) bf16  sXr[2][LC * XRLD];   // 2 x 65 KB
  __shared__ alignas(16) bf16  sXc[2][CD * XCLD];   // 2 x 72 KB
  __shared__ alignas(16) float sS[YT * SLD];        // 17 KB
  __shared__ alignas(16) bf16  sP[YT * PLD];        // 9 KB
  __shared__ float sM4[256], sS4[256];
  __shared__ float sRM[YT], sRS[YT], sC[YT], sO[YT];

  const int tid  = threadIdx.x;
  const int wave = tid >> 5;
  const int lane = tid & 31;
  const int y0   = blockIdx.x * YT;
  const int b    = blockIdx.y;
  const int mi    = wave & 3;          // 16-row M strip of the label tile
  const int whalf = wave >> 2;         // half of N (GEMM1) / of D (GEMM2)
  const int m = lane & 15, h = lane >> 4;

  // ---- U tile A-fragments resident in registers (reused for 40 chunks) ---
  v16bf aU[16];
  {
    const bf16* ur = Ub + (size_t)(y0 + mi * 16 + m) * CD;
#pragma unroll
    for (int ks = 0; ks < 16; ++ks) {
      union { uint4 u[2]; v16bf v; } t;
      t.u[0] = *(const uint4*)(ur + ks * 32 + 8 * h);
      t.u[1] = *(const uint4*)(ur + ks * 32 + 16 + 8 * h);
      aU[ks] = t.v;
    }
  }
  if (tid < YT) { sRM[tid] = -INFINITY; sRS[tid] = 0.0f; }

  v8f mAcc[16];
#pragma unroll
  for (int t = 0; t < 16; ++t) mAcc[t] = zero8();

  const bf16* xrb = Xr + (size_t)b * LPAD * CD;
  const bf16* xtb = Xt + (size_t)b * CD * LPAD;
  constexpr int nCh = (CL + LC - 1) / LC;   // 40

  stage_chunk(xrb, xtb, 0, sXr[0], sXc[0], tid);   // prologue: chunk 0

  for (int ch = 0; ch < nCh; ++ch) {
    const int cur = ch & 1;
    const int l0 = ch * LC;
    const int lv = (CL - l0) < LC ? (CL - l0) : LC;

    // prefetch next chunk into the other buffer, then wait for current
    if (ch + 1 < nCh) {
      stage_chunk(xrb, xtb, (ch + 1) * LC, sXr[cur ^ 1], sXc[cur ^ 1], tid);
      WAIT_ASYNC(32);
    } else {
      WAIT_ASYNC(0);
    }
    __syncthreads();                               // (C) chunk data visible

    // ---- GEMM1: S(64x64) = U_tile @ Xc^T ---------------------------------
    v8f sAcc[2];
    sAcc[0] = zero8(); sAcc[1] = zero8();
#pragma unroll
    for (int ks = 0; ks < CD / 32; ++ks) {
#pragma unroll
      for (int j = 0; j < 2; ++j) {
        const v16bf bb =
            loadB(sXr[cur], XRLD, (whalf * 2 + j) * 16, ks * 32, lane);
        sAcc[j] = __builtin_amdgcn_wmma_f32_16x16x32_bf16(
            false, aU[ks], false, bb, (short)0, sAcc[j], false, false);
      }
    }
    // spill S tiles (C/D layout: lane n=lane&15, VGPR v -> M = v + 8h)
#pragma unroll
    for (int j = 0; j < 2; ++j) {
      const int ni = whalf * 2 + j;
#pragma unroll
      for (int v = 0; v < 8; ++v)
        sS[(mi * 16 + v + 8 * h) * SLD + ni * 16 + m] = sAcc[j][v];
    }
    __syncthreads();                               // (E1)

    // ---- online softmax: 4 threads per label row -------------------------
    {
      const int r = tid >> 2, q = tid & 3;
      const float om = sRM[r];
      float lm = -INFINITY;
#pragma unroll
      for (int j = q * 16; j < q * 16 + 16; ++j)
        if (j < lv) lm = fmaxf(lm, sS[r * SLD + j]);
      sM4[tid] = lm;
      __syncthreads();                             // (E2)
      const float nm =
          fmaxf(om, fmaxf(fmaxf(sM4[r * 4 + 0], sM4[r * 4 + 1]),
                          fmaxf(sM4[r * 4 + 2], sM4[r * 4 + 3])));
      float psum = 0.0f;
#pragma unroll
      for (int j = q * 16; j < q * 16 + 16; ++j) {
        if (j < lv) {
          const float e = __expf(sS[r * SLD + j] - nm);
          psum += e;
          sP[r * PLD + j] = f2bf(e);
        } else {
          sP[r * PLD + j] = f2bf(0.0f);
        }
      }
      sS4[tid] = psum;
      __syncthreads();                             // (E3)
      if (q == 0) {
        const float tot =
            sS4[r * 4 + 0] + sS4[r * 4 + 1] + sS4[r * 4 + 2] + sS4[r * 4 + 3];
        const float corr = __expf(om - nm);        // 0 on first chunk
        sRS[r] = sRS[r] * corr + tot;
        sRM[r] = nm;
        sC[r]  = corr;
      }
    }
    __syncthreads();                               // (E4)

    // ---- rescale running context + GEMM2: m += P @ Xc --------------------
    {
      float c8[8];
#pragma unroll
      for (int v = 0; v < 8; ++v) c8[v] = sC[mi * 16 + v + 8 * h];
#pragma unroll
      for (int t = 0; t < 16; ++t)
#pragma unroll
        for (int v = 0; v < 8; ++v) mAcc[t][v] *= c8[v];

#pragma unroll
      for (int ks = 0; ks < LC / 32; ++ks) {
        const v16bf a = loadA(sP, PLD, mi * 16, ks * 32, lane);
#pragma unroll
        for (int t = 0; t < 16; ++t) {
          const v16bf bb =
              loadB(sXc[cur], XCLD, whalf * 256 + t * 16, ks * 32, lane);
          mAcc[t] = __builtin_amdgcn_wmma_f32_16x16x32_bf16(
              false, a, false, bb, (short)0, mAcc[t], false, false);
        }
      }
    }
    __syncthreads();                               // (G) buf[cur] free
  }

  // ---- epilogue: y = (sum_d m*final_w)/rowsum + final_b -------------------
  if (tid < YT) sO[tid] = 0.0f;
  __syncthreads();
  {
    float part[8];
#pragma unroll
    for (int v = 0; v < 8; ++v) part[v] = 0.0f;
#pragma unroll
    for (int t = 0; t < 16; ++t) {
      const int d = whalf * 256 + t * 16 + m;
#pragma unroll
      for (int v = 0; v < 8; ++v) {
        const int y = y0 + mi * 16 + v + 8 * h;
        const float w = (y < CY) ? final_w[y * CD + d] : 0.0f;
        part[v] += mAcc[t][v] * w;
      }
    }
#pragma unroll
    for (int v = 0; v < 8; ++v)
      atomicAdd(&sO[mi * 16 + v + 8 * h], part[v]);
  }
  __syncthreads();
  if (tid < YT) {
    const int y = y0 + tid;
    if (y < CY) out[b * CY + y] = sO[tid] / sRS[tid] + final_b[y];
  }
}

// ---------------------------------------------------------------------------
extern "C" void kernel_launch(void* const* d_in, const int* in_sizes, int n_in,
                              void* d_out, int out_size, void* d_ws,
                              size_t ws_size, hipStream_t stream) {
  (void)in_sizes; (void)n_in; (void)out_size; (void)ws_size;
  const float* x       = (const float*)d_in[0];
  // d_in[1] = target (int32) -- unused by the forward pass
  const float* U_w     = (const float*)d_in[2];
  const float* final_w = (const float*)d_in[3];
  const float* final_b = (const float*)d_in[4];
  float* out = (float*)d_out;

  // workspace layout (needs ~30.2 MB)
  const size_t ub_bytes = (size_t)YPAD * CD * sizeof(bf16);      //  9,175,040
  const size_t xr_bytes = (size_t)CB * LPAD * CD * sizeof(bf16); // 10,485,760
  bf16* Ub = (bf16*)d_ws;
  bf16* Xr = (bf16*)((char*)d_ws + ub_bytes);
  bf16* Xt = (bf16*)((char*)d_ws + ub_bytes + xr_bytes);

  // pre-conversion passes
  prep_U<<<(YPAD * CD / 4) / 256, 256, 0, stream>>>(U_w, Ub);
  dim3 gx(LPAD / 32, CD / 32, CB);
  prep_x<<<gx, 256, 0, stream>>>(x, Xr, Xt);

  // fused attention
  dim3 grid(YPAD / YT, CB);
  laat_main<<<grid, 256, 0, stream>>>(Ub, Xr, Xt, final_w, final_b, out);
}
